// PointNet_Plus_58188216926612
// MI455X (gfx1250) — matrix operand order repaired and checked by reading, not compiled, not run on hardware
//
#include <hip/hip_runtime.h>
#include <hip/hip_bf16.h>

typedef __attribute__((ext_vector_type(16))) __bf16 v16bf;
typedef __attribute__((ext_vector_type(8)))  float    v8f;
typedef __attribute__((ext_vector_type(8)))  unsigned v8u;
typedef __attribute__((ext_vector_type(4)))  unsigned v4u;

__device__ __forceinline__ __bf16 f2bf(float f) {
  unsigned u = __builtin_bit_cast(unsigned, f);
  unsigned r = u + 0x7FFFu + ((u >> 16) & 1u);
  unsigned short h = (unsigned short)(r >> 16);
  return __builtin_bit_cast(__bf16, h);
}
__device__ __forceinline__ unsigned pk2(float lo, float hi) {
  unsigned ul = (unsigned)__builtin_bit_cast(unsigned short, f2bf(lo));
  unsigned uh = (unsigned)__builtin_bit_cast(unsigned short, f2bf(hi));
  return ul | (uh << 16);
}

__constant__ int c_REL1[16] = {1,0,1,2,0,4,5,0,7,8,0,10,11,0,13,14};
__constant__ int c_REL2[16] = {13,2,3,3,5,6,6,8,9,9,11,12,12,14,15,15};
__constant__ float c_SKEL[32] = {
  0.f,0.15f,0.25f,0.35f,0.21f,0.21f,0.21f,0.f,0.f,0.f,-0.125f,-0.125f,-0.125f,-0.17f,-0.17f,-0.17f,
  0.f,0.f,0.1f,0.2f,0.21f,0.36f,0.46f,0.3f,0.45f,0.55f,0.22f,0.37f,0.47f,0.1f,0.21f,0.3f};

// ---------------------------------------------------------------------------
// Tiled bf16 layouts (one 32B vector = one lane's WMMA operand registers):
//   A tiles:  [mt][ks][lane 0..31][16 bf16],  k = ks*32 + lh*16 + j*2 + e
//   B tiles:  [ks][nt][lane 0..31][16 bf16],  n = nt*16 + lane&15, k = ks*32 + lh*16 + idx
//   D (f32):  n = lane&15, m = vgpr + 8*lh  ->  packs into B-tile slot idx = r + 8*lh
// ---------------------------------------------------------------------------

// WMMA with reuse_a hint on all but the first instruction of an identical chain.
#define WMMA0(A, B, C) __builtin_amdgcn_wmma_f32_16x16x32_bf16(false, A, false, B, (short)0, C, false, false)
#define WMMAR(A, B, C) __builtin_amdgcn_wmma_f32_16x16x32_bf16(false, A, false, B, (short)0, C, true,  false)

// Fused 3-layer 1x1-conv (+folded BN+ReLU) + max over NCOLS columns.
// One wave per group; layers 1/2 staged in per-wave LDS B-tiles.
template<int C1, int C2, int C3, int NCOLS, int WAVES, int KS1>
__global__ __launch_bounds__(WAVES*32)
void fused_mlp3_max(const __bf16* __restrict__ tin,
                    const __bf16* __restrict__ tW1, const float* __restrict__ s1, const float* __restrict__ h1,
                    const __bf16* __restrict__ tW2, const float* __restrict__ s2, const float* __restrict__ h2,
                    const __bf16* __restrict__ tW3, const float* __restrict__ s3, const float* __restrict__ h3,
                    float* __restrict__ out, int ngroups,
                    int ogdiv, long ogout, long ochs, long ogin, long ooff)
{
  static_assert(C1 % 32 == 0 && C2 % 32 == 0 && C3 % 16 == 0 && NCOLS % 16 == 0, "dims");
  constexpr int NT  = NCOLS / 16;
  constexpr int KS2 = C1 / 32;
  constexpr int KS3 = C2 / 32;
  __shared__ v8u smem[WAVES * (C1 + C2) * NCOLS / 16];
  const int lane = threadIdx.x & 31;
  const int wave = threadIdx.x >> 5;
  const int g = blockIdx.x * WAVES + wave;
  if (g >= ngroups) return;
  v8u* bufA = smem + wave * ((C1 + C2) * NCOLS / 16);
  v8u* bufB = bufA + C1 * NCOLS / 16;
  const v8u* tg = (const v8u*)tin + (long)g * (KS1 * NT * 32);
  const v8u* w1 = (const v8u*)tW1;
  const v8u* w2 = (const v8u*)tW2;
  const v8u* w3 = (const v8u*)tW3;
  const int nlo = lane & 15, lh = lane >> 4;

  // ---------------- Layer 1 (global B tiles) ----------------
  for (int mt = 0; mt < C1 / 16; ++mt) {
    v8f acc[NT];
    #pragma unroll
    for (int nt = 0; nt < NT; ++nt)
      #pragma unroll
      for (int r = 0; r < 8; ++r) acc[nt][r] = 0.0f;
    for (int ks = 0; ks < KS1; ++ks) {
      if (ks + 1 < KS1)
        __builtin_prefetch((const void*)(tg + ((ks + 1) * NT) * 32 + lane), 0, 1);
      v16bf a = __builtin_bit_cast(v16bf, w1[(mt * KS1 + ks) * 32 + lane]);
      {
        v16bf b = __builtin_bit_cast(v16bf, tg[(ks * NT + 0) * 32 + lane]);
        acc[0] = WMMA0(a, b, acc[0]);
      }
      #pragma unroll
      for (int nt = 1; nt < NT; ++nt) {
        v16bf b = __builtin_bit_cast(v16bf, tg[(ks * NT + nt) * 32 + lane]);
        acc[nt] = WMMAR(a, b, acc[nt]);
      }
    }
    float sA[8], hA[8];
    #pragma unroll
    for (int r = 0; r < 8; ++r) { int m = mt * 16 + r + (lh << 3); sA[r] = s1[m]; hA[r] = h1[m]; }
    int bl = nlo + ((mt & 1) << 4);
    #pragma unroll
    for (int nt = 0; nt < NT; ++nt) {
      v4u w;
      #pragma unroll
      for (int p = 0; p < 4; ++p) {
        float f0 = fmaxf(acc[nt][2*p]   * sA[2*p]   + hA[2*p],   0.0f);
        float f1 = fmaxf(acc[nt][2*p+1] * sA[2*p+1] + hA[2*p+1], 0.0f);
        w[p] = pk2(f0, f1);
      }
      v4u* dst = (v4u*)(bufA + ((mt >> 1) * NT + nt) * 32 + bl);
      dst[lh] = w;
    }
  }
  // ---------------- Layer 2 (LDS B tiles) ----------------
  for (int mt = 0; mt < C2 / 16; ++mt) {
    v8f acc[NT];
    #pragma unroll
    for (int nt = 0; nt < NT; ++nt)
      #pragma unroll
      for (int r = 0; r < 8; ++r) acc[nt][r] = 0.0f;
    for (int ks = 0; ks < KS2; ++ks) {
      v16bf a = __builtin_bit_cast(v16bf, w2[(mt * KS2 + ks) * 32 + lane]);
      {
        v16bf b = __builtin_bit_cast(v16bf, bufA[(ks * NT + 0) * 32 + lane]);
        acc[0] = WMMA0(a, b, acc[0]);
      }
      #pragma unroll
      for (int nt = 1; nt < NT; ++nt) {
        v16bf b = __builtin_bit_cast(v16bf, bufA[(ks * NT + nt) * 32 + lane]);
        acc[nt] = WMMAR(a, b, acc[nt]);
      }
    }
    float sA[8], hA[8];
    #pragma unroll
    for (int r = 0; r < 8; ++r) { int m = mt * 16 + r + (lh << 3); sA[r] = s2[m]; hA[r] = h2[m]; }
    int bl = nlo + ((mt & 1) << 4);
    #pragma unroll
    for (int nt = 0; nt < NT; ++nt) {
      v4u w;
      #pragma unroll
      for (int p = 0; p < 4; ++p) {
        float f0 = fmaxf(acc[nt][2*p]   * sA[2*p]   + hA[2*p],   0.0f);
        float f1 = fmaxf(acc[nt][2*p+1] * sA[2*p+1] + hA[2*p+1], 0.0f);
        w[p] = pk2(f0, f1);
      }
      v4u* dst = (v4u*)(bufB + ((mt >> 1) * NT + nt) * 32 + bl);
      dst[lh] = w;
    }
  }
  // ---------------- Layer 3 + max over columns ----------------
  for (int mt = 0; mt < C3 / 16; ++mt) {
    v8f acc[NT];
    #pragma unroll
    for (int nt = 0; nt < NT; ++nt)
      #pragma unroll
      for (int r = 0; r < 8; ++r) acc[nt][r] = 0.0f;
    for (int ks = 0; ks < KS3; ++ks) {
      v16bf a = __builtin_bit_cast(v16bf, w3[(mt * KS3 + ks) * 32 + lane]);
      {
        v16bf b = __builtin_bit_cast(v16bf, bufB[(ks * NT + 0) * 32 + lane]);
        acc[0] = WMMA0(a, b, acc[0]);
      }
      #pragma unroll
      for (int nt = 1; nt < NT; ++nt) {
        v16bf b = __builtin_bit_cast(v16bf, bufB[(ks * NT + nt) * 32 + lane]);
        acc[nt] = WMMAR(a, b, acc[nt]);
      }
    }
    float sA[8], hA[8];
    #pragma unroll
    for (int r = 0; r < 8; ++r) { int m = mt * 16 + r + (lh << 3); sA[r] = s3[m]; hA[r] = h3[m]; }
    v8f mx;
    #pragma unroll
    for (int r = 0; r < 8; ++r) mx[r] = -3.0e38f;
    #pragma unroll
    for (int nt = 0; nt < NT; ++nt)
      #pragma unroll
      for (int r = 0; r < 8; ++r)
        mx[r] = fmaxf(mx[r], fmaxf(acc[nt][r] * sA[r] + hA[r], 0.0f));
    #pragma unroll
    for (int off = 1; off < 16; off <<= 1)
      #pragma unroll
      for (int r = 0; r < 8; ++r) mx[r] = fmaxf(mx[r], __shfl_xor(mx[r], off, 32));
    if (nlo == 0) {
      long ob = (long)(g / ogdiv) * ogout + (long)(g % ogdiv) * ogin + ooff;
      #pragma unroll
      for (int r = 0; r < 8; ++r)
        out[ob + (long)(mt * 16 + r + (lh << 3)) * ochs] = mx[r];
    }
  }
}

// ---------------------------------------------------------------------------
// Packing kernels
// ---------------------------------------------------------------------------
// fp32 row-major W[Cout][Cin] -> bf16 A-tiles [mt][ks][lane][16], zero pad K.
__global__ void weight_pack_kernel(const float* __restrict__ W, __bf16* __restrict__ dst,
                                   int Cout, int Cin, int KS) {
  int t = blockIdx.x * blockDim.x + threadIdx.x;
  int total = (Cout / 16) * KS * 32;
  if (t >= total) return;
  int lane = t & 31;
  int ks = (t >> 5) % KS;
  int mt = (t >> 5) / KS;
  int m = mt * 16 + (lane & 15);
  int lh = lane >> 4;
  const float* wr = W + (long)m * Cin;
  v8u o;
  #pragma unroll
  for (int j = 0; j < 8; ++j) {
    int k0 = ks * 32 + (lh << 4) + (j << 1);
    float f0 = (k0     < Cin) ? wr[k0]     : 0.0f;
    float f1 = (k0 + 1 < Cin) ? wr[k0 + 1] : 0.0f;
    o[j] = pk2(f0, f1);
  }
  ((v8u*)dst)[t] = o;
}

// Generic fp32 source (strided channels) -> bf16 B-tiles [g][ks][nt][lane][16].
__global__ void input_pack_kernel(const float* __restrict__ src, __bf16* __restrict__ dst,
                                  int CIN, int KS, int NT, int ngroups,
                                  int gdiv, long gout, long gin, long chs) {
  int t = blockIdx.x * blockDim.x + threadIdx.x;
  int total = ngroups * KS * NT * 32;
  if (t >= total) return;
  int lane = t & 31;
  int nt = (t >> 5) % NT;
  int ks = ((t >> 5) / NT) % KS;
  int g = t / (32 * NT * KS);
  const float* gi = src + (long)(g / gdiv) * gout + (long)(g % gdiv) * gin;
  int n = nt * 16 + (lane & 15);
  int kb = ks * 32 + ((lane >> 4) << 4);
  v8u o;
  #pragma unroll
  for (int j = 0; j < 8; ++j) {
    int k0 = kb + 2 * j;
    float f0 = (k0     < CIN) ? gi[(long)k0 * chs + n]       : 0.0f;
    float f1 = (k0 + 1 < CIN) ? gi[(long)(k0 + 1) * chs + n] : 0.0f;
    o[j] = pk2(f0, f1);
  }
  ((v8u*)dst)[t] = o;
}

// ---------------------------------------------------------------------------
// Support kernels
// ---------------------------------------------------------------------------
__global__ void bn_fold_kernel(const float* b, const float* g, const float* be,
                               const float* mu, const float* var,
                               float* sc, float* sh, int C) {
  int i = blockIdx.x * blockDim.x + threadIdx.x;
  if (i >= C) return;
  float s = g[i] * rsqrtf(var[i] + 1e-5f);
  sc[i] = s;
  sh[i] = (b[i] - mu[i]) * s + be[i];
}

__global__ void y_to_level1_kernel(const float* y, float* level1) {
  int i = blockIdx.x * blockDim.x + threadIdx.x;
  if (i >= 32 * 3 * 512) return;
  int s = i & 511, c = (i >> 9) % 3, b = i / (3 * 512);
  level1[((long)b * 131 + c) * 512 + s] = y[i];
}

__global__ void ctr_to_l2_kernel(const float* level1, float* l2) {
  int i = blockIdx.x * blockDim.x + threadIdx.x;
  if (i >= 32 * 3 * 128) return;
  int s = i & 127, c = (i >> 7) % 3, b = i / 384;
  l2[((long)b * 259 + c) * 128 + s] = level1[((long)b * 131 + c) * 512 + s];
}

// top-64 (ascending d2, stable ties) + radius fallback to self + gather,
// written directly as bf16 B-tiles [g][5][4][lane][16] (131 ch padded to 160).
__global__ void group2_kernel(const float* __restrict__ level1, __bf16* __restrict__ tiled) {
  int b = blockIdx.x >> 7, c = blockIdx.x & 127;
  __shared__ float sd[512];
  __shared__ int sidx[64];
  const float* L = level1 + (long)b * 131 * 512;
  float cx = L[c], cy = L[512 + c], cz = L[1024 + c];
  for (int p = threadIdx.x; p < 512; p += blockDim.x) {
    float dx = L[p] - cx, dy = L[512 + p] - cy, dz = L[1024 + p] - cz;
    sd[p] = dx * dx + dy * dy + dz * dz;
  }
  __syncthreads();
  for (int p = threadIdx.x; p < 512; p += blockDim.x) {
    float dp = sd[p];
    int rank = 0;
    for (int q = 0; q < 512; ++q) {
      float dq = sd[q];
      rank += (dq < dp || (dq == dp && q < p)) ? 1 : 0;
    }
    if (rank < 64) sidx[rank] = p;
  }
  __syncthreads();
  __bf16* dst = tiled + (long)blockIdx.x * (5 * 4 * 512);
  for (int t = threadIdx.x; t < 160 * 64; t += blockDim.x) {
    int q = t >> 6, k = t & 63;
    float v = 0.0f;
    if (q < 131) {
      int id = sidx[k];
      if (sd[id] > 0.04f) id = c;
      v = L[(long)q * 512 + id];
      if (q < 3) v -= L[(long)q * 512 + c];
    }
    int off = ((q >> 5) * 4 + (k >> 4)) * 512 + ((k & 15) + (((q >> 4) & 1) << 4)) * 16 + (q & 15);
    dst[off] = f2bf(v);
  }
}

// final_group + const features (fold/feat/rel_a/rel_b), emitted as bf16 B-tiles
// [g][KS][4][lane][16] with CIN channels zero-padded to KS*32.
__global__ void stage_prep_kernel(const float* __restrict__ level1, const float* __restrict__ fold,
                                  const float* __restrict__ feat, int cf,
                                  __bf16* __restrict__ tiled, int split, int CIN, int KS) {
  int b = blockIdx.x >> 4, j = blockIdx.x & 15;
  __shared__ float sd[512];
  __shared__ int sidx[64];
  const float* L = level1 + (long)b * 131 * 512;
  float fx = fold[((long)b * 3 + 0) * 16 + j];
  float fy = fold[((long)b * 3 + 1) * 16 + j];
  float fz = fold[((long)b * 3 + 2) * 16 + j];
  for (int p = threadIdx.x; p < 512; p += blockDim.x) {
    float dx = L[p] - fx, dy = L[512 + p] - fy, dz = L[1024 + p] - fz;
    sd[p] = dx * dx + dy * dy + dz * dz;
  }
  __syncthreads();
  for (int p = threadIdx.x; p < 512; p += blockDim.x) {
    float dp = sd[p];
    int rank = 0;
    for (int q = 0; q < 512; ++q) {
      float dq = sd[q];
      rank += (dq < dp || (dq == dp && q < p)) ? 1 : 0;
    }
    if (rank < 64) sidx[rank] = p;
  }
  __syncthreads();
  __bf16* dst = tiled + (long)blockIdx.x * ((long)KS * 4 * 512);
  int padded = KS * 32;
  for (int t = threadIdx.x; t < padded * 64; t += blockDim.x) {
    int q = t >> 6, k = t & 63;
    float v = 0.0f;
    if (q < 3) {
      v = fold[((long)b * 3 + q) * 16 + j];
    } else if (q < 3 + cf) {
      v = feat[((long)b * cf + (q - 3)) * 16 + j];
    } else if (q < 3 + 2 * cf) {
      v = feat[((long)b * cf + (q - 3 - cf)) * 16 + c_REL1[j]];
    } else if (q < split) {
      v = feat[((long)b * cf + (q - 3 - 2 * cf)) * 16 + c_REL2[j]];
    } else if (q < CIN) {
      int qq = q - split;
      int id = sidx[k];
      if (sd[id] > 0.16f) id = sidx[0];
      v = L[(long)qq * 512 + id];
      if (qq < 3) v -= fold[((long)b * 3 + qq) * 16 + j];
    }
    int off = ((q >> 5) * 4 + (k >> 4)) * 512 + ((k & 15) + (((q >> 4) & 1) << 4)) * 16 + (q & 15);
    dst[off] = f2bf(v);
  }
}

// Scalar 1x1-conv layer for tiny N=16 fold MLPs. sc==nullptr -> plain conv+bias.
__global__ void scalar_layer_kernel(const float* __restrict__ inp, const float* __restrict__ W,
                                    const float* __restrict__ sc, const float* __restrict__ sh,
                                    float* __restrict__ out, int Cin, int Cout, int N, int relu) {
  int i = blockIdx.x * blockDim.x + threadIdx.x;
  int total = 32 * Cout * N;
  if (i >= total) return;
  int n = i % N, m = (i / N) % Cout, b = i / (N * Cout);
  const float* a = inp + (long)b * Cin * N + n;
  const float* w = W + (long)m * Cin;
  float acc = 0.0f;
  for (int c = 0; c < Cin; ++c) acc += w[c] * a[(long)c * N];
  float s = sc ? sc[m] : 1.0f;
  float v = acc * s + sh[m];
  if (relu) v = fmaxf(v, 0.0f);
  out[i] = v;
}

__global__ void build_fin_kernel(const float* code, float* fin) {
  int i = blockIdx.x * blockDim.x + threadIdx.x;
  if (i >= 32 * 514 * 16) return;
  int j = i & 15, c = (i >> 4) % 514, b = i / (514 * 16);
  fin[i] = (c < 2) ? c_SKEL[c * 16 + j] : code[(long)b * 512 + (c - 2)];
}

__global__ void fold_out_kernel(const float* conv, const float* prev, float* foldbuf, float* dout) {
  int i = blockIdx.x * blockDim.x + threadIdx.x;
  if (i >= 1536) return;
  float v = conv[i] + (prev ? prev[i] : 0.0f);
  foldbuf[i] = v;
  int j = i & 15, c = (i >> 4) % 3, b = i / 48;
  dout[(long)b * 48 + j * 3 + c] = v;
}

// ---------------------------------------------------------------------------
extern "C" void kernel_launch(void* const* d_in, const int* in_sizes, int n_in,
                              void* d_out, int out_size, void* d_ws, size_t ws_size,
                              hipStream_t stream) {
  (void)in_sizes; (void)n_in; (void)out_size; (void)ws_size;
  auto in = [&](int i) { return (const float*)d_in[i]; };
  const float* x = in(0);
  const float* y = in(1);
  enum { F11 = 2, F12 = 20, F21 = 22, F22M = 40, F22O = 52,
         F31 = 54, F32M = 72, F32O = 84, R1 = 86, R2 = 104, R3 = 122 };

  float* ws = (float*)d_ws;
  size_t off = 0;
  auto alloc = [&](size_t n) -> float* { float* p = ws + off; off += (n + 63) & ~(size_t)63; return p; };

  auto bnfold = [&](int base, int C, float** sc, float** sh) {
    *sc = alloc(C); *sh = alloc(C);
    bn_fold_kernel<<<dim3((C + 63) / 64), dim3(64), 0, stream>>>(
        in(base + 1), in(base + 2), in(base + 3), in(base + 4), in(base + 5), *sc, *sh, C);
  };
  float *sc_r1[3], *sh_r1[3], *sc_r2[3], *sh_r2[3], *sc_r3[3], *sh_r3[3];
  float *sc_f11[3], *sh_f11[3], *sc_f21[3], *sh_f21[3], *sc_f31[3], *sh_f31[3];
  float *sc_f22[2], *sh_f22[2], *sc_f32[2], *sh_f32[2];
  { int c[3] = {32, 32, 128};   for (int l = 0; l < 3; ++l) bnfold(R1  + 6 * l, c[l], &sc_r1[l],  &sh_r1[l]); }
  { int c[3] = {64, 64, 256};   for (int l = 0; l < 3; ++l) bnfold(R2  + 6 * l, c[l], &sc_r2[l],  &sh_r2[l]); }
  { int c[3] = {128, 128, 512}; for (int l = 0; l < 3; ++l) bnfold(R3  + 6 * l, c[l], &sc_r3[l],  &sh_r3[l]); }
  { int c[3] = {256, 256, 128}; for (int l = 0; l < 3; ++l) bnfold(F11 + 6 * l, c[l], &sc_f11[l], &sh_f11[l]); }
  { int c[3] = {256, 256, 256}; for (int l = 0; l < 3; ++l) bnfold(F21 + 6 * l, c[l], &sc_f21[l], &sh_f21[l]); }
  { int c[2] = {256, 256};      for (int l = 0; l < 2; ++l) bnfold(F22M+ 6 * l, c[l], &sc_f22[l], &sh_f22[l]); }
  { int c[3] = {256, 256, 256}; for (int l = 0; l < 3; ++l) bnfold(F31 + 6 * l, c[l], &sc_f31[l], &sh_f31[l]); }
  { int c[2] = {256, 256};      for (int l = 0; l < 2; ++l) bnfold(F32M+ 6 * l, c[l], &sc_f32[l], &sh_f32[l]); }

  // Pre-tiled bf16 weights (A-layout).
  auto wpack = [&](const float* W, int Cout, int Cin) -> __bf16* {
    int KS = (Cin + 31) / 32;
    __bf16* t = (__bf16*)alloc((size_t)Cout * KS * 16);
    int total = (Cout / 16) * KS * 32;
    weight_pack_kernel<<<dim3((total + 255) / 256), dim3(256), 0, stream>>>(W, t, Cout, Cin, KS);
    return t;
  };
  __bf16* tw_r1[3] = { wpack(in(R1), 32, 6),    wpack(in(R1 + 6), 32, 32),   wpack(in(R1 + 12), 128, 32) };
  __bf16* tw_r2[3] = { wpack(in(R2), 64, 131),  wpack(in(R2 + 6), 64, 64),   wpack(in(R2 + 12), 256, 64) };
  __bf16* tw_r3[3] = { wpack(in(R3), 128, 259), wpack(in(R3 + 6), 128, 128), wpack(in(R3 + 12), 512, 128) };
  __bf16* tw_f21[3]= { wpack(in(F21), 256, 518),wpack(in(F21 + 6), 256, 256),wpack(in(F21 + 12), 256, 256) };
  __bf16* tw_f31[3]= { wpack(in(F31), 256, 902),wpack(in(F31 + 6), 256, 256),wpack(in(F31 + 12), 256, 256) };

  float* level1 = alloc(32L * 131 * 512);
  float* l2     = alloc(32L * 259 * 128);
  float* code   = alloc(32L * 512);
  // Big tiled region, reused sequentially: x-tiles (33.5M) -> act2-tiles (41.9M) -> stage-tiles (30.4M)
  __bf16* big   = (__bf16*)alloc(4096L * 5 * 4 * 512 / 2);
  __bf16* tl2   = (__bf16*)alloc(32L * 9 * 8 * 512 / 2);
  float* fin    = alloc(32L * 514 * 16);
  float* fold11 = alloc(32L * 128 * 16);
  float* feat2  = alloc(32L * 256 * 16);
  float* feat3  = alloc(32L * 256 * 16);
  float* fold1b = alloc(1536);
  float* fold2b = alloc(1536);
  float* fold3b = alloc(1536);
  float* ta     = alloc(32L * 256 * 16);
  float* tb     = alloc(32L * 256 * 16);
  float* tc     = alloc(1536);
  float* dout   = (float*)d_out;

  auto slayer = [&](const float* inp, const float* W, const float* sc, const float* sh,
                    float* outp, int Cin, int Cout, int N, int relu) {
    int total = 32 * Cout * N;
    scalar_layer_kernel<<<dim3((total + 255) / 256), dim3(256), 0, stream>>>(
        inp, W, sc, sh, outp, Cin, Cout, N, relu);
  };

  // ---- encoder stage 1 (r1) ----
  y_to_level1_kernel<<<dim3(192), dim3(256), 0, stream>>>(y, level1);
  input_pack_kernel<<<dim3((16384 * 1 * 4 * 32 + 255) / 256), dim3(256), 0, stream>>>(
      x, big, 6, 1, 4, 16384, 512, 6L * 512 * 64, 64L, 512L * 64);
  fused_mlp3_max<32, 32, 128, 64, 4, 1><<<dim3(4096), dim3(128), 0, stream>>>(
      big, tw_r1[0], sc_r1[0], sh_r1[0], tw_r1[1], sc_r1[1], sh_r1[1], tw_r1[2], sc_r1[2], sh_r1[2],
      level1, 16384, 512, 131L * 512, 512L, 1L, 3L * 512);

  // ---- grouping + gather, tiled ----
  group2_kernel<<<dim3(4096), dim3(128), 0, stream>>>(level1, big);

  // ---- encoder stage 2 (r2) ----
  ctr_to_l2_kernel<<<dim3(48), dim3(256), 0, stream>>>(level1, l2);
  fused_mlp3_max<64, 64, 256, 64, 2, 5><<<dim3(2048), dim3(64), 0, stream>>>(
      big, tw_r2[0], sc_r2[0], sh_r2[0], tw_r2[1], sc_r2[1], sh_r2[1], tw_r2[2], sc_r2[2], sh_r2[2],
      l2, 4096, 128, 259L * 128, 128L, 1L, 3L * 128);

  // ---- encoder stage 3 (r3) ----
  input_pack_kernel<<<dim3((32 * 9 * 8 * 32 + 255) / 256), dim3(256), 0, stream>>>(
      l2, tl2, 259, 9, 8, 32, 1, 259L * 128, 0L, 128L);
  fused_mlp3_max<128, 128, 512, 128, 1, 9><<<dim3(32), dim3(32), 0, stream>>>(
      tl2, tw_r3[0], sc_r3[0], sh_r3[0], tw_r3[1], sc_r3[1], sh_r3[1], tw_r3[2], sc_r3[2], sh_r3[2],
      code, 32, 1, 512L, 1L, 0L, 0L);

  // ---- fold 1 ----
  build_fin_kernel<<<dim3((32 * 514 * 16 + 255) / 256), dim3(256), 0, stream>>>(code, fin);
  slayer(fin,    in(F11),      sc_f11[0], sh_f11[0], ta,     514, 256, 16, 1);
  slayer(ta,     in(F11 + 6),  sc_f11[1], sh_f11[1], tb,     256, 256, 16, 1);
  slayer(tb,     in(F11 + 12), sc_f11[2], sh_f11[2], fold11, 256, 128, 16, 1);
  slayer(fold11, in(F12),      nullptr,   in(F12 + 1), tc,   128, 3,   16, 0);
  fold_out_kernel<<<dim3(6), dim3(256), 0, stream>>>(tc, nullptr, fold1b, dout);

  // ---- stage 2 ----
  stage_prep_kernel<<<dim3(512), dim3(128), 0, stream>>>(level1, fold1b, fold11, 128, big, 387, 518, 17);
  fused_mlp3_max<256, 256, 256, 64, 1, 17><<<dim3(512), dim3(32), 0, stream>>>(
      big, tw_f21[0], sc_f21[0], sh_f21[0], tw_f21[1], sc_f21[1], sh_f21[1], tw_f21[2], sc_f21[2], sh_f21[2],
      feat2, 512, 16, 256L * 16, 16L, 1L, 0L);
  slayer(feat2, in(F22M),     sc_f22[0], sh_f22[0], ta, 256, 256, 16, 1);
  slayer(ta,    in(F22M + 6), sc_f22[1], sh_f22[1], tb, 256, 256, 16, 1);
  slayer(tb,    in(F22O),     nullptr,   in(F22O + 1), tc, 256, 3, 16, 0);
  fold_out_kernel<<<dim3(6), dim3(256), 0, stream>>>(tc, fold1b, fold2b, dout + 1536);

  // ---- stage 3 ----
  stage_prep_kernel<<<dim3(512), dim3(128), 0, stream>>>(level1, fold2b, feat2, 256, big, 771, 902, 29);
  fused_mlp3_max<256, 256, 256, 64, 1, 29><<<dim3(512), dim3(32), 0, stream>>>(
      big, tw_f31[0], sc_f31[0], sh_f31[0], tw_f31[1], sc_f31[1], sh_f31[1], tw_f31[2], sc_f31[2], sh_f31[2],
      feat3, 512, 16, 256L * 16, 16L, 1L, 0L);
  slayer(feat3, in(F32M),     sc_f32[0], sh_f32[0], ta, 256, 256, 16, 1);
  slayer(ta,    in(F32M + 6), sc_f32[1], sh_f32[1], tb, 256, 256, 16, 1);
  slayer(tb,    in(F32O),     nullptr,   in(F32O + 1), tc, 256, 3, 16, 0);
  fold_out_kernel<<<dim3(6), dim3(256), 0, stream>>>(tc, fold2b, fold3b, dout + 3072);
}
